// OfflineSlidingWindowAttn_5325759447104
// MI455X (gfx1250) — matrix operand order repaired and checked by reading, not compiled, not run on hardware
//
#include <hip/hip_runtime.h>
#include <hip/hip_bf16.h>

typedef __attribute__((ext_vector_type(16))) _Float16 v16h;
typedef __attribute__((ext_vector_type(8)))  _Float16 v8h;
typedef __attribute__((ext_vector_type(8)))  float    v8f;

#define HEAD_DIM   128
#define NUM_Q_HEAD 32
#define NUM_KV_HEAD 8
#define REP        4
#define SEQ        2048
#define BATCH      2
#define WINDOW     512

// LDS strides (in halfs), padded to avoid bank conflicts
#define KSTR 136   // KN rows: 128 + 8
#define VSTR 552   // VT rows: 544 + 8
#define QSTR 136
#define PSTR 40

static constexpr float F_EPS   = 1e-5f;
static constexpr float F_SCALE = 0.08838834764831845f;  // 1/sqrt(128)
static constexpr float F_CAP   = 50.0f;
static constexpr float F_CA    = 1.02f;                 // CLIP_R - CLIP_L
static constexpr float F_CB    = -0.01f;                // CLIP_L

#if __has_builtin(__builtin_amdgcn_rcpf)
#define RCPF(x) __builtin_amdgcn_rcpf(x)
#else
#define RCPF(x) (1.0f / (x))
#endif

// exp(CAP*tanh(s*SCALE/CAP) - CAP), fully branch-free.
// tanh(x) = 1 - 2/(e^{2x}+1)  =>  result = exp(-2*CAP / (e^{2*s*SCALE/CAP} + 1)).
// |s*SCALE/CAP| <= ~0.23 for RMS-normalized inputs, so no range issues.
__device__ __forceinline__ float cap_exp(float s) {
#if __has_builtin(__builtin_amdgcn_tanhf)
    const float th = __builtin_amdgcn_tanhf(s * (F_SCALE / F_CAP));
    return __expf(F_CAP * th - F_CAP);
#else
    const float e2 = __expf(s * (2.0f * F_SCALE / F_CAP));
    return __expf((-2.0f * F_CAP) * RCPF(e2 + 1.0f));
#endif
}

__device__ __forceinline__ v16h pack16(v8h lo, v8h hi) {
    v16h r;
#pragma unroll
    for (int i = 0; i < 8; ++i) { r[i] = lo[i]; r[i + 8] = hi[i]; }
    return r;
}

__device__ __forceinline__ v8f wmma_f16(v16h a, v16h b, v8f c) {
    // D = A(16x32 f16) * B(32x16 f16) + C(16x16 f32)
    return __builtin_amdgcn_wmma_f32_16x16x32_f16(false, a, false, b, (short)0, c, false, false);
}

__global__ __launch_bounds__(128, 1)
void OfflineSlidingWindowAttn_kernel(const float* __restrict__ q,
                                     const float* __restrict__ k,
                                     const float* __restrict__ v,
                                     const float* __restrict__ gq,
                                     const float* __restrict__ gk,
                                     float* __restrict__ out) {
    extern __shared__ char smem_raw[];
    _Float16* KN = (_Float16*)smem_raw;        // [544][KSTR]  normalized K, f16
    _Float16* VT = KN + 544 * KSTR;            // [128][VSTR]  V transposed, f16
    _Float16* QN = VT + 128 * VSTR;            // [4][16][QSTR] normalized Q, f16
    _Float16* PB = QN + 4 * 16 * QSTR;         // [4][16][PSTR] prob staging

    const int wg  = blockIdx.x;
    const int qt  = wg & 127;                  // SEQ/16 = 128 q tiles
    const int kvh = (wg >> 7) & 7;
    const int bb  = wg >> 10;
    const int q0  = qt * 16;
    const int kstart = (q0 - WINDOW) > 0 ? (q0 - WINDOW) : 0;
    const int nkeys  = q0 + 16 - kstart;       // multiple of 16, <= 528
    const int chunks = (nkeys + 31) >> 5;      // 32-key chunks, <= 17

    const int tid  = threadIdx.x;
    const int wave = tid >> 5;
    const int lane = tid & 31;
    const int half = lane >> 4;
    const int col  = lane & 15;
    const int qh   = kvh * REP + wave;

    // ---------------- cooperative preload: K (normed), V (transposed), Q (normed) -------------
    {
        const int g  = tid >> 2;   // 0..31: row within a 32-row pass
        const int dq = tid & 3;    // 0..3 : 32-element d-quarter
        float gkv[32];
#pragma unroll
        for (int e = 0; e < 32; ++e) gkv[e] = gk[kvh * HEAD_DIM + dq * 32 + e];

        for (int it = 0; it < chunks; ++it) {
            const int kk = it * 32 + g;
            const int jj = kstart + kk;
            const bool valid = (kk < nkeys);
            float kx[32], vx[32];
            if (valid) {
                const size_t base = (((size_t)bb * SEQ + jj) * NUM_KV_HEAD + kvh) * HEAD_DIM + dq * 32;
                const float4* ks = (const float4*)(k + base);
                const float4* vs = (const float4*)(v + base);
#pragma unroll
                for (int e = 0; e < 8; ++e) {
                    float4 a = ks[e], bvv = vs[e];
                    kx[4*e+0]=a.x;  kx[4*e+1]=a.y;  kx[4*e+2]=a.z;  kx[4*e+3]=a.w;
                    vx[4*e+0]=bvv.x; vx[4*e+1]=bvv.y; vx[4*e+2]=bvv.z; vx[4*e+3]=bvv.w;
                }
            } else {
#pragma unroll
                for (int e = 0; e < 32; ++e) { kx[e] = 0.f; vx[e] = 0.f; }
            }
            float ss = 0.f;
#pragma unroll
            for (int e = 0; e < 32; ++e) ss += kx[e] * kx[e];
            ss += __shfl_xor(ss, 1, 32);
            ss += __shfl_xor(ss, 2, 32);
            const float inv = valid ? (1.0f / sqrtf(ss * (1.0f / 128.0f) + F_EPS)) : 0.f;

            _Float16* kd = KN + kk * KSTR + dq * 32;
#pragma unroll
            for (int e8 = 0; e8 < 4; ++e8) {
                v8h h;
#pragma unroll
                for (int i = 0; i < 8; ++i) h[i] = (_Float16)(kx[e8*8+i] * inv * gkv[e8*8+i]);
                *(v8h*)(kd + e8 * 8) = h;
            }
#pragma unroll
            for (int e = 0; e < 32; ++e)
                VT[(dq * 32 + e) * VSTR + kk] = (_Float16)vx[e];
        }

        // Q: 64 rows (4 heads x 16 rows), 32 rows per pass
#pragma unroll
        for (int it = 0; it < 2; ++it) {
            const int r   = it * 32 + g;
            const int hl  = r >> 4;
            const int qr  = r & 15;
            const int qh2 = kvh * REP + hl;
            float qx[32];
            const size_t base = (((size_t)bb * SEQ + q0 + qr) * NUM_Q_HEAD + qh2) * HEAD_DIM + dq * 32;
            const float4* qs = (const float4*)(q + base);
#pragma unroll
            for (int e = 0; e < 8; ++e) {
                float4 a = qs[e];
                qx[4*e+0]=a.x; qx[4*e+1]=a.y; qx[4*e+2]=a.z; qx[4*e+3]=a.w;
            }
            float ss = 0.f;
#pragma unroll
            for (int e = 0; e < 32; ++e) ss += qx[e] * qx[e];
            ss += __shfl_xor(ss, 1, 32);
            ss += __shfl_xor(ss, 2, 32);
            const float inv = 1.0f / sqrtf(ss * (1.0f / 128.0f) + F_EPS);
            _Float16* qd = QN + (hl * 16 + qr) * QSTR + dq * 32;
#pragma unroll
            for (int e8 = 0; e8 < 4; ++e8) {
                v8h h;
#pragma unroll
                for (int i = 0; i < 8; ++i)
                    h[i] = (_Float16)(qx[e8*8+i] * inv * gq[qh2 * HEAD_DIM + dq * 32 + e8*8 + i]);
                *(v8h*)(qd + e8 * 8) = h;
            }
        }
    }
    __syncthreads();

    // ---------------- per-wave: A fragments of Q (K=128 -> 4 chunks of 32) ---------------------
    v16h aq[4];
    {
        const _Float16* qrow = QN + (wave * 16 + col) * QSTR;
#pragma unroll
        for (int c = 0; c < 4; ++c) {
            v8h lo = *(const v8h*)(qrow + 32 * c + 8 * half);
            v8h hi = *(const v8h*)(qrow + 32 * c + 16 + 8 * half);
            aq[c] = pack16(lo, hi);
        }
    }

    // ---------------- pass 1: row denominators with fixed max = CAP ---------------------------
    float linv[8];
#pragma unroll
    for (int r = 0; r < 8; ++r) linv[r] = 0.f;

    for (int ch = 0; ch < chunks; ++ch) {
        const int kb = ch * 32;
        v8f s0 = {0,0,0,0,0,0,0,0}, s1 = {0,0,0,0,0,0,0,0};
#pragma unroll
        for (int c = 0; c < 4; ++c) {
            const _Float16* kr0 = KN + (kb + col) * KSTR + 32 * c + 8 * half;
            const _Float16* kr1 = kr0 + 16 * KSTR;
            v16h b0 = pack16(*(const v8h*)kr0, *(const v8h*)(kr0 + 16));
            v16h b1 = pack16(*(const v8h*)kr1, *(const v8h*)(kr1 + 16));
            s0 = wmma_f16(aq[c], b0, s0);
            s1 = wmma_f16(aq[c], b1, s1);
        }
        const int j0 = kstart + kb + col;
        const int j1 = j0 + 16;
#pragma unroll
        for (int r = 0; r < 8; ++r) {
            const int i = q0 + 8 * half + r;
            const bool a0 = (j0 <= i) && (i - j0 <= WINDOW);
            const bool a1 = (j1 <= i) && (i - j1 <= WINDOW);
            const float e0 = a0 ? cap_exp(s0[r]) : 0.f;
            const float e1 = a1 ? cap_exp(s1[r]) : 0.f;
            linv[r] += e0 + e1;
        }
    }
#pragma unroll
    for (int r = 0; r < 8; ++r) {
        float x = linv[r];
        x += __shfl_xor(x, 1, 32);
        x += __shfl_xor(x, 2, 32);
        x += __shfl_xor(x, 4, 32);
        x += __shfl_xor(x, 8, 32);
        linv[r] = 1.0f / x;
    }

    // ---------------- pass 2: recompute S, clip probs, P@V ------------------------------------
    v8f acc[8];
#pragma unroll
    for (int d = 0; d < 8; ++d) acc[d] = (v8f){0,0,0,0,0,0,0,0};
    _Float16* pw = PB + wave * 16 * PSTR;

    for (int ch = 0; ch < chunks; ++ch) {
        const int kb = ch * 32;
        v8f s0 = {0,0,0,0,0,0,0,0}, s1 = {0,0,0,0,0,0,0,0};
#pragma unroll
        for (int c = 0; c < 4; ++c) {
            const _Float16* kr0 = KN + (kb + col) * KSTR + 32 * c + 8 * half;
            const _Float16* kr1 = kr0 + 16 * KSTR;
            v16h b0 = pack16(*(const v8h*)kr0, *(const v8h*)(kr0 + 16));
            v16h b1 = pack16(*(const v8h*)kr1, *(const v8h*)(kr1 + 16));
            s0 = wmma_f16(aq[c], b0, s0);
            s1 = wmma_f16(aq[c], b1, s1);
        }
        const int j0 = kstart + kb + col;
        const int j1 = j0 + 16;
        __syncthreads();   // PB reuse guard (all waves run identical trip counts)
#pragma unroll
        for (int r = 0; r < 8; ++r) {
            const int i = q0 + 8 * half + r;
            const bool a0 = (j0 <= i) && (i - j0 <= WINDOW);
            const bool a1 = (j1 <= i) && (i - j1 <= WINDOW);
            const float e0 = a0 ? cap_exp(s0[r]) : 0.f;
            const float e1 = a1 ? cap_exp(s1[r]) : 0.f;
            const float p0 = fminf(fmaxf(F_CA * e0 * linv[r] + F_CB, 0.f), 1.f);
            const float p1 = fminf(fmaxf(F_CA * e1 * linv[r] + F_CB, 0.f), 1.f);
            pw[(8 * half + r) * PSTR + col]      = (_Float16)p0;
            pw[(8 * half + r) * PSTR + 16 + col] = (_Float16)p1;
        }
        __syncthreads();   // cross-lane P visibility
        v16h pa;
        {
            const _Float16* prow = pw + col * PSTR;
            v8h lo = *(const v8h*)(prow + 8 * half);
            v8h hi = *(const v8h*)(prow + 16 + 8 * half);
            pa = pack16(lo, hi);
        }
#pragma unroll
        for (int dt = 0; dt < 8; ++dt) {
            const _Float16* vr = VT + (16 * dt + col) * VSTR + kb + 8 * half;
            v16h bv = pack16(*(const v8h*)vr, *(const v8h*)(vr + 16));
            acc[dt] = wmma_f16(pa, bv, acc[dt]);
        }
    }

    // ---------------- write O: [B, S, Hq, D] fp32 ---------------------------------------------
#pragma unroll
    for (int dt = 0; dt < 8; ++dt)
#pragma unroll
        for (int r = 0; r < 8; ++r) {
            const int i = q0 + 8 * half + r;
            out[(((size_t)bb * SEQ + i) * NUM_Q_HEAD + qh) * HEAD_DIM + 16 * dt + col] = acc[dt][r];
        }
}

extern "C" void kernel_launch(void* const* d_in, const int* in_sizes, int n_in,
                              void* d_out, int out_size, void* d_ws, size_t ws_size,
                              hipStream_t stream) {
    (void)in_sizes; (void)n_in; (void)out_size; (void)d_ws; (void)ws_size;
    const float* q  = (const float*)d_in[0];
    const float* k  = (const float*)d_in[1];
    const float* v  = (const float*)d_in[2];
    const float* gq = (const float*)d_in[3];
    const float* gk = (const float*)d_in[4];
    float* out = (float*)d_out;

    const int blocks = BATCH * NUM_KV_HEAD * (SEQ / 16);   // 2048
    const size_t shmem = (size_t)(544 * KSTR + 128 * VSTR + 4 * 16 * QSTR + 4 * 16 * PSTR)
                         * sizeof(_Float16);               // ~305 KB < 320 KB/WGP
    OfflineSlidingWindowAttn_kernel<<<blocks, 128, shmem, stream>>>(q, k, v, gq, gk, out);
}
